// BasePllay_05_2_42992622633500
// MI455X (gfx1250) — compile-verified, matrix operand order checked by reading, not run on hardware
//
#include <hip/hip_runtime.h>

// ---------------- constants ----------------
#define HH 224
#define WW 224
#define NPIX (HH * WW)           // 50176
#define BB 2
#define TT 25
#define MPTS 64
#define OUTF 50

#define R0 37
#define SIDE0 (2 * R0 + 1)       // 75
#define NOFF0 (SIDE0 * SIDE0)    // 5625
#define R1 74
#define SIDE1 (2 * R1 + 1)       // 149
#define NOFF1 (SIDE1 * SIDE1)    // 22201

#define TILES1 4                 // N tiles for 50 -> 64
#define STEPS1 2                 // K steps for 50 -> 64  (K=32 per WMMA)
#define STEPS2 10                // K steps for 300 -> 320

typedef float    v8f  __attribute__((ext_vector_type(8)));
typedef _Float16 v16h __attribute__((ext_vector_type(16)));

struct CfgPtrs {
    const float* tseq[6];
    const float* gw[6];
    const float* gb[6];
};

// Fragment element -> K mapping for CDNA5 16-bit A/B (16x32 / 32x16):
// lane L: hi = L>>4, mn = L&15;  element e -> k = k0 + 8*hi + (e<8 ? e : e+8)
__device__ __host__ inline int frag_k(int step, int hi, int e) {
    return 32 * step + 8 * hi + (e < 8 ? e : e + 8);
}

// ---------------- 1) offset tables: stable counting sort by d2 ----------------
__global__ void setup_offsets_kernel(int r, int noff, int* opack, float* od2, int* hist) {
    if (threadIdx.x != 0 || blockIdx.x != 0) return;
    const int side = 2 * r + 1;
    const int d2max = 2 * r * r;
    for (int i = 0; i <= d2max; ++i) hist[i] = 0;
    for (int idx = 0; idx < noff; ++idx) {
        int dy = idx / side - r, dx = idx % side - r;
        hist[dy * dy + dx * dx]++;
    }
    int run = 0;
    for (int i = 0; i <= d2max; ++i) { int h = hist[i]; hist[i] = run; run += h; }
    for (int idx = 0; idx < noff; ++idx) {          // idx ascending -> stable
        int dy = idx / side - r, dx = idx % side - r;
        int d2 = dy * dy + dx * dx;
        int pos = hist[d2]++;
        opack[pos] = ((dy + r) << 16) | (dx + r);
        od2[pos] = (float)d2;
    }
}

// ---------------- 2) per-image weight sums ----------------
__global__ void sum_kernel(const float* in, float* sums) {
    __shared__ float s[256];
    const int img = blockIdx.x;                      // b*3 + c
    const float* p = in + (size_t)img * NPIX;
    float acc = 0.f;
    for (int i = threadIdx.x; i < NPIX; i += 256) acc += p[i];
    s[threadIdx.x] = acc;
    __syncthreads();
    for (int st = 128; st > 0; st >>= 1) {
        if ((int)threadIdx.x < st) s[threadIdx.x] += s[threadIdx.x + st];
        __syncthreads();
    }
    if (threadIdx.x == 0) sums[img] = s[0];
}

// ---------------- 3) DTM: one thread per pixel, early-exit cumsum walk ----------------
__global__ void dtm_kernel(const float* in, const float* sums,
                           const int* __restrict__ opack, const float* __restrict__ od2,
                           int noff, int r, float m0, float* dt_out) {
    int gid = blockIdx.x * blockDim.x + threadIdx.x;
    if (gid >= BB * 3 * NPIX) return;
    int p = gid % NPIX;
    int bc = gid / NPIX;
    int b = bc / 3, c = bc % 3;
    const float* w = in + (size_t)(b * 3 + c) * NPIX;
    const float thr = m0 * sums[b * 3 + c];
    const int y = p / WW, x = p % WW;
    float cum = 0.f, acc = 0.f;
    for (int o = 0; o < noff; ++o) {
        if (cum >= thr) break;
        if ((o & 63) == 0) {
            __builtin_prefetch(opack + o + 256, 0, 1);
            __builtin_prefetch(od2 + o + 256, 0, 1);
        }
        int pk = opack[o];
        int dy = (pk >> 16) - r, dx = (pk & 0xffff) - r;
        int yy = y + dy, xx = x + dx;
        float val = 0.f;
        if ((unsigned)yy < (unsigned)HH && (unsigned)xx < (unsigned)WW) val = w[yy * WW + xx];
        float rem = thr - cum;                        // exclusive cumsum
        float used = fminf(fmaxf(rem, 0.f), val);
        acc += used * od2[o];
        cum += val;
    }
    dt_out[((size_t)(c * BB + b)) * NPIX + p] = sqrtf(fmaxf(acc / thr, 0.f));
}

// ---------------- 4) H0 persistence: serial per image (12 independent tasks) ----------------
__device__ inline int uf_find(int* parent, int i) {
    while (parent[i] != i) { parent[i] = parent[parent[i]]; i = parent[i]; }
    return i;
}

__global__ void persistence_kernel(const float* dt,
                                   unsigned long long* kvA, unsigned long long* kvB,
                                   int* parentBuf, int* pyBuf, int* pdBuf, float* ppBuf,
                                   float* bdBuf) {
    if (threadIdx.x != 0) return;
    const int img = blockIdx.x;                      // cfg*2 + b
    const float* f = dt + (size_t)img * NPIX;
    unsigned long long* A  = kvA + (size_t)img * NPIX;
    unsigned long long* Bt = kvB + (size_t)img * NPIX;
    int*   parent = parentBuf + (size_t)img * NPIX;
    int*   py = pyBuf + (size_t)img * NPIX;
    int*   pd = pdBuf + (size_t)img * NPIX;
    float* pp = ppBuf + (size_t)img * NPIX;
    float* bd = bdBuf + (size_t)img * 2 * MPTS;

    // stable argsort: key = (float bits of dt) : (index); dt >= 0 so bits monotonic
    for (int i = 0; i < NPIX; ++i)
        A[i] = ((unsigned long long)__float_as_uint(f[i]) << 32) | (unsigned)i;
    unsigned long long* src = A;
    unsigned long long* dst = Bt;
    for (int pass = 0; pass < 4; ++pass) {           // LSD radix over high 32 bits
        int shift = 32 + pass * 8;
        int hist[256];
        for (int i = 0; i < 256; ++i) hist[i] = 0;
        for (int i = 0; i < NPIX; ++i) hist[(int)((src[i] >> shift) & 255)]++;
        int run = 0;
        for (int i = 0; i < 256; ++i) { int h = hist[i]; hist[i] = run; run += h; }
        for (int i = 0; i < NPIX; ++i) dst[hist[(int)((src[i] >> shift) & 255)]++] = src[i];
        unsigned long long* t = src; src = dst; dst = t;
    }                                                 // 4 passes -> result back in A (== src)

    for (int i = 0; i < NPIX; ++i) parent[i] = -1;
    int npairs = 0;
    for (int s = 0; s < NPIX; ++s) {
        int i = (int)(src[s] & 0xffffffffull);
        parent[i] = i;
        int y = i / WW, x = i - y * WW;
        int nbs[4]; int nn = 0;
        if (y > 0)      nbs[nn++] = i - WW;
        if (y < HH - 1) nbs[nn++] = i + WW;
        if (x > 0)      nbs[nn++] = i - 1;
        if (x < WW - 1) nbs[nn++] = i + 1;
        for (int q = 0; q < nn; ++q) {
            int nb = nbs[q];
            if (parent[nb] < 0) continue;
            int ri = uf_find(parent, i);
            int rn = uf_find(parent, nb);
            if (ri == rn) continue;
            int elder, young;
            if (f[ri] <= f[rn]) { elder = ri; young = rn; } else { elder = rn; young = ri; }
            py[npairs] = young; pd[npairs] = i; pp[npairs] = f[i] - f[young]; ++npairs;
            parent[young] = elder;
        }
    }
    int bm = 0, bM = 0;                               // np.argmin / np.argmax (first occurrence)
    for (int i = 1; i < NPIX; ++i) {
        if (f[i] < f[bm]) bm = i;
        if (f[i] > f[bM]) bM = i;
    }
    py[npairs] = bm; pd[npairs] = bM; pp[npairs] = f[bM] - f[bm]; ++npairs;

    // top-64 by persistence, stable (first occurrence of max == argsort(-pers) order)
    int cnt = npairs < MPTS ? npairs : MPTS;
    for (int j = 0; j < MPTS; ++j) {
        if (j < cnt) {
            int best = 0; float bp = -3.4e38f;
            for (int q = 0; q < npairs; ++q)
                if (pp[q] > bp) { bp = pp[q]; best = q; }
            float birth = f[py[best]];
            float death = (bp > 0.f) ? f[pd[best]] : birth;   // mask = pers > 0
            bd[2 * j] = birth; bd[2 * j + 1] = death;
            pp[best] = -3.4e38f;
        } else {
            bd[2 * j] = 0.f; bd[2 * j + 1] = 0.f;            // tent == 0 for t > 0
        }
    }
}

// ---------------- 5) tents + top-2 -> feat (B, K*T) per cfg ----------------
__global__ void feature_kernel(const float* bdBuf, float* feat, CfgPtrs cp) {
    const int img = blockIdx.x;                      // cfg*2 + b
    const int t = threadIdx.x;
    if (t >= TT) return;
    const int cfg = img >> 1;
    const float* bd = bdBuf + (size_t)img * 2 * MPTS;
    const float tv = cp.tseq[cfg][t];
    float l1 = 0.f, l2 = 0.f;
    for (int j = 0; j < MPTS; ++j) {
        float v = fmaxf(fminf(tv - bd[2 * j], bd[2 * j + 1] - tv), 0.f);
        if (v > l1) { l2 = l1; l1 = v; }
        else if (v > l2) { l2 = v; }
    }
    feat[(size_t)img * (2 * TT) + t]      = l1;      // k=0 row
    feat[(size_t)img * (2 * TT) + TT + t] = l2;      // k=1 row
}

// ---------------- 6a) pack gemm1 operands into WMMA fragment order (f16, zero-padded) ----
// A frag (per cfg, step, lane): 16 halves contiguous.  B frag (per cfg, tile, step, lane).
__global__ void pack_gemm1_kernel(const float* feat, CfgPtrs cp,
                                  _Float16* a1, _Float16* b1) {
    const int cfg = blockIdx.x;
    const float* F  = feat + (size_t)cfg * 2 * OUTF;  // [b][k]
    const float* gw = cp.gw[cfg];                     // gw[o*50 + k]
    const int tid = threadIdx.x;                      // 0..255
    if (tid < STEPS1 * 32) {                          // A: (step, lane)
        int step = tid >> 5, lane = tid & 31;
        int hi = lane >> 4, mn = lane & 15;
        _Float16* dst = a1 + (((size_t)cfg * STEPS1 + step) * 32 + lane) * 16;
#pragma unroll
        for (int e = 0; e < 16; ++e) {
            int k = frag_k(step, hi, e);
            dst[e] = (_Float16)((mn < BB && k < OUTF) ? F[mn * OUTF + k] : 0.f);
        }
    }
    {                                                 // B: (tile, step, lane)
        int tile = tid >> 6, step = (tid >> 5) & 1, lane = tid & 31;
        int hi = lane >> 4, mn = lane & 15;
        int o = tile * 16 + mn;
        _Float16* dst = b1 + ((((size_t)cfg * TILES1 + tile) * STEPS1 + step) * 32 + lane) * 16;
#pragma unroll
        for (int e = 0; e < 16; ++e) {
            int k = frag_k(step, hi, e);
            dst[e] = (_Float16)((o < OUTF && k < OUTF) ? gw[o * OUTF + k] : 0.f);
        }
    }
}

// ---------------- 6b) per-cfg GEMM: feat(2x50) @ gw.T(50x50) + gb  (WMMA) ----------------
__global__ void gemm1_kernel(const _Float16* __restrict__ a1, const _Float16* __restrict__ b1,
                             float* xbuf, CfgPtrs cp) {
    const int cfg = blockIdx.x;
    const int lane = threadIdx.x & 31;
    const float* gb = cp.gb[cfg];
    const v16h* Ap = (const v16h*)(a1 + (size_t)cfg * STEPS1 * 32 * 16);
    for (int tile = 0; tile < TILES1; ++tile) {
        const v16h* Bp = (const v16h*)(b1 + (((size_t)cfg * TILES1 + tile) * STEPS1) * 32 * 16);
        v8f acc = {};
#pragma unroll
        for (int s = 0; s < STEPS1; ++s) {
            v16h a = Ap[s * 32 + lane];
            v16h b = Bp[s * 32 + lane];
            acc = __builtin_amdgcn_wmma_f32_16x16x32_f16(false, a, false, b,
                                                         (short)0, acc, false, false);
        }
        if (lane < 16) {
            int o = tile * 16 + lane;
            if (o < OUTF) {
                xbuf[0 * 300 + cfg * OUTF + o] = acc[0] + gb[o];
                xbuf[1 * 300 + cfg * OUTF + o] = acc[1] + gb[o];
            }
        }
    }
}

// ---------------- 7a) pack FC operands (relu(x) and fc_w) ----------------
__global__ void pack_fc_kernel(const float* xbuf, const float* fc_w,
                               _Float16* a2, _Float16* b2) {
    const int t = threadIdx.x;                        // 0..319
    const int step = t >> 5, lane = t & 31;
    const int hi = lane >> 4, mn = lane & 15;
    if (blockIdx.x == 0) {                            // A = relu(x) (2x300 -> 16x320)
        _Float16* dst = a2 + ((size_t)(step * 32 + lane)) * 16;
#pragma unroll
        for (int e = 0; e < 16; ++e) {
            int k = frag_k(step, hi, e);
            float v = (mn < BB && k < 300) ? fmaxf(xbuf[mn * 300 + k], 0.f) : 0.f;
            dst[e] = (_Float16)v;
        }
    } else {                                          // B = fc_w (7x300 -> 16x320)
        _Float16* dst = b2 + ((size_t)(step * 32 + lane)) * 16;
#pragma unroll
        for (int e = 0; e < 16; ++e) {
            int k = frag_k(step, hi, e);
            float v = (mn < 7 && k < 300) ? fc_w[mn * 300 + k] : 0.f;
            dst[e] = (_Float16)v;
        }
    }
}

// ---------------- 7b) final: relu(x)(2x300) @ fc_w.T(300x7) + fc_b ; signal ----------------
__global__ void final_kernel(const _Float16* __restrict__ a2, const _Float16* __restrict__ b2,
                             const float* xbuf, const float* fc_b, float* out) {
    if (threadIdx.x < 32) {                           // wave 0: WMMA FC (uniform branch)
        const int lane = threadIdx.x;
        const v16h* Ap = (const v16h*)a2;
        const v16h* Bp = (const v16h*)b2;
        v8f acc = {};
        for (int s = 0; s < STEPS2; ++s) {
            v16h a = Ap[s * 32 + lane];
            v16h b = Bp[s * 32 + lane];
            acc = __builtin_amdgcn_wmma_f32_16x16x32_f16(false, a, false, b,
                                                         (short)0, acc, false, false);
        }
        if (lane < 7) {
            out[lane]     = acc[0] + fc_b[lane];
            out[7 + lane] = acc[1] + fc_b[lane];
        }
    } else {                                          // other waves: signal = sum_b |x|
        int k = (int)threadIdx.x - 32;
        if (k < 300) out[14 + k] = fabsf(xbuf[k]) + fabsf(xbuf[300 + k]);
    }
}

// ---------------- host: workspace layout + launches ----------------
extern "C" void kernel_launch(void* const* d_in, const int* in_sizes, int n_in,
                              void* d_out, int out_size, void* d_ws, size_t ws_size,
                              hipStream_t stream) {
    (void)in_sizes; (void)n_in; (void)out_size; (void)ws_size;
    const float* input = (const float*)d_in[0];
    CfgPtrs cp;
    for (int cfg = 0; cfg < 6; ++cfg) {
        cp.tseq[cfg] = (const float*)d_in[1 + 3 * cfg];
        cp.gw[cfg]   = (const float*)d_in[2 + 3 * cfg];
        cp.gb[cfg]   = (const float*)d_in[3 + 3 * cfg];
    }
    const float* fc_w = (const float*)d_in[19];
    const float* fc_b = (const float*)d_in[20];
    float* out = (float*)d_out;

    char* base = (char*)d_ws;
    size_t off = 0;
    auto alloc = [&](size_t bytes) -> char* {
        char* p = base + off;
        off = (off + bytes + 63) & ~(size_t)63;       // 64B align (v16h needs 32B)
        return p;
    };
    int*   opack0 = (int*)  alloc(NOFF0 * 4);
    float* od2_0  = (float*)alloc(NOFF0 * 4);
    int*   opack1 = (int*)  alloc(NOFF1 * 4);
    float* od2_1  = (float*)alloc(NOFF1 * 4);
    int*   hist0  = (int*)  alloc(11008 * 4);
    int*   hist1  = (int*)  alloc(11008 * 4);
    float* sums   = (float*)alloc(6 * 4);
    float* dt     = (float*)alloc((size_t)6 * BB * NPIX * 4);
    unsigned long long* kvA = (unsigned long long*)alloc((size_t)12 * NPIX * 8);
    unsigned long long* kvB = (unsigned long long*)alloc((size_t)12 * NPIX * 8);
    int*   parentB = (int*)  alloc((size_t)12 * NPIX * 4);
    int*   pyB     = (int*)  alloc((size_t)12 * NPIX * 4);
    int*   pdB     = (int*)  alloc((size_t)12 * NPIX * 4);
    float* ppB     = (float*)alloc((size_t)12 * NPIX * 4);
    float* bdB     = (float*)alloc((size_t)12 * 2 * MPTS * 4);
    float* feat    = (float*)alloc((size_t)6 * BB * 2 * TT * 4);
    float* xbuf    = (float*)alloc((size_t)BB * 300 * 4);
    _Float16* a1   = (_Float16*)alloc((size_t)6 * STEPS1 * 32 * 16 * 2);           // 12 KB
    _Float16* b1   = (_Float16*)alloc((size_t)6 * TILES1 * STEPS1 * 32 * 16 * 2);  // 48 KB
    _Float16* a2   = (_Float16*)alloc((size_t)STEPS2 * 32 * 16 * 2);               // 10 KB
    _Float16* b2   = (_Float16*)alloc((size_t)STEPS2 * 32 * 16 * 2);               // 10 KB

    setup_offsets_kernel<<<1, 1, 0, stream>>>(R0, NOFF0, opack0, od2_0, hist0);
    setup_offsets_kernel<<<1, 1, 0, stream>>>(R1, NOFF1, opack1, od2_1, hist1);
    sum_kernel<<<6, 256, 0, stream>>>(input, sums);

    const int total = BB * 3 * NPIX;
    const int blocks = (total + 255) / 256;
    dtm_kernel<<<blocks, 256, 0, stream>>>(input, sums, opack0, od2_0,
                                           NOFF0, R0, 0.05f, dt);
    dtm_kernel<<<blocks, 256, 0, stream>>>(input, sums, opack1, od2_1,
                                           NOFF1, R1, 0.2f, dt + (size_t)3 * BB * NPIX);

    persistence_kernel<<<12, 32, 0, stream>>>(dt, kvA, kvB, parentB, pyB, pdB, ppB, bdB);
    feature_kernel<<<12, 32, 0, stream>>>(bdB, feat, cp);
    pack_gemm1_kernel<<<6, 256, 0, stream>>>(feat, cp, a1, b1);
    gemm1_kernel<<<6, 32, 0, stream>>>(a1, b1, xbuf, cp);
    pack_fc_kernel<<<2, 320, 0, stream>>>(xbuf, fc_w, a2, b2);
    final_kernel<<<1, 352, 0, stream>>>(a2, b2, xbuf, fc_b, out);
}
